// Attention_12532714569861
// MI455X (gfx1250) — matrix operand order repaired and checked by reading, not compiled, not run on hardware
//
#include <hip/hip_runtime.h>

// ---- problem constants (match reference) ----
#define B_  2
#define Q_  2048
#define K_  2048
#define H_  16
#define D_  64
#define DV_ 64
#define NKT (K_ / 32)          // 64 chunks of 32 keys

#define LOG2E      1.4426950408889634f
#define SCALE_L2E  (0.125f * LOG2E)     // 1/sqrt(64) * log2(e)
#define MASKVAL    (-1.0e9f * LOG2E)    // NEG_INF in log2 domain
#define NEG_BIG    (-3.0e38f)
#define KB_STRIDE  72                    // padded halves per K-tile row (144B, bank-spread)
#define VT_STRIDE  40                    // padded halves per dv-row of V^T tile (80B)

typedef __attribute__((ext_vector_type(16))) _Float16 v16h;
typedef __attribute__((ext_vector_type(8)))  _Float16 v8h;
typedef __attribute__((ext_vector_type(8)))  float    v8f;
typedef __attribute__((ext_vector_type(4)))  float    f32x4;

static __device__ __forceinline__ v8f wmma_f16(v16h a, v16h b, v8f c) {
  // D = A(16x32 f16) * B(32x16 f16) + C(16x16 f32)
  return __builtin_amdgcn_wmma_f32_16x16x32_f16(
      false, a, false, b, (short)0, c, false, false);
}

static __device__ __forceinline__ v16h cat8(v8h lo, v8h hi) {
  return __builtin_shufflevector(lo, hi, 0,1,2,3,4,5,6,7,8,9,10,11,12,13,14,15);
}

static __device__ __forceinline__ v8h cvt8(f32x4 a, f32x4 b) {
  v8h h;
  h[0] = (_Float16)a.x; h[1] = (_Float16)a.y; h[2] = (_Float16)a.z; h[3] = (_Float16)a.w;
  h[4] = (_Float16)b.x; h[5] = (_Float16)b.y; h[6] = (_Float16)b.z; h[7] = (_Float16)b.w;
  return h;
}

// ---- CDNA5 async global->LDS copy (ASYNCcnt-tracked, ISA 15.18.3 op 98) ----
static __device__ __forceinline__ void async_b128(const _Float16* g, const _Float16* l) {
  const unsigned long long ga = (unsigned long long)(uintptr_t)g;
  const unsigned int       la = (unsigned int)(uintptr_t)l;   // flat LDS addr[31:0] = LDS offset
  asm volatile("global_load_async_to_lds_b128 %0, %1, off"
               :: "v"(la), "v"(ga) : "memory");
}
static __device__ __forceinline__ void wait_async0() {
  asm volatile("s_wait_asynccnt 0x0" ::: "memory");
}

// each thread moves one 16B chunk: 256 threads == one 32x64 f16 K tile (padded rows)
static __device__ __forceinline__ void stage_k_async(const _Float16* khf, int c32, int tid,
                                                     _Float16* kb) {
  const int row = tid >> 3;            // 0..31 key row
  const int seg = (tid & 7) * 8;       // halves within row
  async_b128(khf + (size_t)(c32 * 32 + row) * (H_ * D_) + seg,
             kb + row * KB_STRIDE + seg);
}
// one 64(dv) x 32(k) f16 V^T tile (padded rows)
static __device__ __forceinline__ void stage_v_async(const _Float16* vtf, int c32, int tid,
                                                     _Float16* vt) {
  const int dv = tid >> 2;             // 0..63
  const int ks = (tid & 3) * 8;        // halves within k-row
  async_b128(vtf + (size_t)dv * K_ + c32 * 32 + ks,
             vt + dv * VT_STRIDE + ks);
}

// B-fragment for S = Q*K^T from row-major f16 K tile in LDS (subtile j)
static __device__ __forceinline__ v16h kfrag(const _Float16* kb, int lane, int j, int d0) {
  const int n   = lane & 15;
  const int seg = (lane >> 4) * 8;
  const _Float16* p = kb + (j * 16 + n) * KB_STRIDE + d0 + seg;   // 16B aligned
  return cat8(*(const v8h*)p, *(const v8h*)(p + 16));
}
// B-fragment for P*V from transposed f16 V tile in LDS
static __device__ __forceinline__ v16h vfrag(const _Float16* vt, int lane, int cch) {
  const int dv = cch * 16 + (lane & 15);
  const _Float16* p = vt + dv * VT_STRIDE + (lane >> 4) * 8;      // 16B aligned
  return cat8(*(const v8h*)p, *(const v8h*)(p + 16));
}

// A-fragment (16x32 f16) for Q, loaded once per wave from global f32
static __device__ __forceinline__ v16h load_qfrag(const float* qrow, int lane, int d0) {
  const int based = (lane >> 4) * 8;
  const f32x4* p0 = (const f32x4*)(qrow + d0 + based);
  const f32x4* p1 = (const f32x4*)(qrow + d0 + based + 16);
  return cat8(cvt8(p0[0], p0[1]), cvt8(p1[0], p1[1]));
}

// ================= prologue kernel 1: K f32 -> f16 (same layout) =================
__global__ __launch_bounds__(256) void cvt_k_kernel(const float* __restrict__ src,
                                                    _Float16* __restrict__ dst) {
  const size_t i = ((size_t)blockIdx.x * 256 + threadIdx.x) * 8;
  const f32x4* g = (const f32x4*)(src + i);
  *(v8h*)(dst + i) = cvt8(g[0], g[1]);
}

// ================= prologue kernel 2: V f32 [b,k,h,dv] -> f16 V^T [b,h,dv,k] =====
__global__ __launch_bounds__(256) void transpose_v_kernel(const float* __restrict__ vs,
                                                          _Float16* __restrict__ vtws) {
  __shared__ _Float16 t[64 * VT_STRIDE];
  const int tid = threadIdx.x;
  const int blk = blockIdx.x;                 // [b][h][kt]
  const int kt  = blk & (K_ / 32 - 1);
  const int h   = (blk >> 6) & (H_ - 1);
  const int b   = blk >> 10;
  const int k0  = kt * 32;
  // phase 1: coalesced f32 reads, scattered f16 transposed stores into LDS
  {
    const int i = tid * 8, k = i >> 6, dv = i & 63;
    const float* g = vs + (((size_t)(b * K_ + k0 + k) * H_) + h) * DV_ + dv;
    f32x4 a = *(const f32x4*)g;
    f32x4 c = *(const f32x4*)(g + 4);
    t[(dv + 0) * VT_STRIDE + k] = (_Float16)a.x;
    t[(dv + 1) * VT_STRIDE + k] = (_Float16)a.y;
    t[(dv + 2) * VT_STRIDE + k] = (_Float16)a.z;
    t[(dv + 3) * VT_STRIDE + k] = (_Float16)a.w;
    t[(dv + 4) * VT_STRIDE + k] = (_Float16)c.x;
    t[(dv + 5) * VT_STRIDE + k] = (_Float16)c.y;
    t[(dv + 6) * VT_STRIDE + k] = (_Float16)c.z;
    t[(dv + 7) * VT_STRIDE + k] = (_Float16)c.w;
  }
  __syncthreads();
  // phase 2: coalesced 16B f16 writes of the transposed tile
  {
    const int dv = tid >> 2, ks = (tid & 3) * 8;
    v8h h8 = *(const v8h*)(t + dv * VT_STRIDE + ks);
    *(v8h*)(vtws + ((size_t)(b * H_ + h) * DV_ + dv) * K_ + k0 + ks) = h8;
  }
}

// ================= main attention kernel =========================================
__global__ __launch_bounds__(256) void attn_fwd_kernel(
    const float*    __restrict__ qs,  const _Float16* __restrict__ kf,
    const _Float16* __restrict__ vtf, const float*    __restrict__ bias,
    const int*      __restrict__ valid_lens,
    float* __restrict__ ctx, float* __restrict__ attn)
{
  __shared__ _Float16 kbuf[2][32 * KB_STRIDE];   // 9 KB : K chunk, padded row-major f16
  __shared__ _Float16 vtbuf[2][64 * VT_STRIDE];  // 10 KB: V^T chunk, padded f16
  __shared__ _Float16 pbuf[8][16 * 32];          // 8 KB : per-wave P re-layout

  const int tid     = threadIdx.x;
  const int lane    = tid & 31;
  const int wid     = tid >> 5;
  const int gw      = blockIdx.x * 8 + wid;      // all 8 waves share (b,h)
  const int qt      = gw & 127;
  const int h       = (gw >> 7) & 15;
  const int b       = gw >> 11;
  const int q0      = qt * 16;
  const int n       = lane & 15;
  const int halfsel = lane >> 4;
  const int vlen    = valid_lens[b];

  const float* qrow = qs + (((size_t)(b * Q_ + q0 + n) * H_) + h) * D_;
  const v16h a0 = load_qfrag(qrow, lane, 0);
  const v16h a1 = load_qfrag(qrow, lane, 32);

  const _Float16* khf = kf  + ((size_t)b * K_ * H_ + h) * D_;   // f16 [k][h][d] slice
  const _Float16* vhf = vtf + (size_t)(b * H_ + h) * DV_ * K_;  // f16 [dv][k] slice
  const float* bias_q = bias + (size_t)(q0 + 8 * halfsel) * K_; // + r*K_ + kc

  // ================= Pass 1: softmax stats (per-lane online, log2 domain) =========
  float m8[8], l8[8];
#pragma unroll
  for (int r = 0; r < 8; ++r) { m8[r] = NEG_BIG; l8[r] = 0.0f; }

  stage_k_async(khf, 0, tid, kbuf[0]);
  for (int c = 0; c < NKT; ++c) {
    wait_async0();                                 // my async writes landed
    __syncthreads();                               // everyone's landed; prev readers done
    if (c + 1 < NKT) stage_k_async(khf, c + 1, tid, kbuf[(c + 1) & 1]);
    const _Float16* kb = kbuf[c & 1];
    __builtin_prefetch(bias_q + c * 32 + 32 + n, 0, 1);  // gfx1250 global_prefetch_b8
#pragma unroll
    for (int j = 0; j < 2; ++j) {
      v8f s8 = {};
      s8 = wmma_f16(a0, kfrag(kb, lane, j, 0),  s8);
      s8 = wmma_f16(a1, kfrag(kb, lane, j, 32), s8);
      const int  kc     = c * 32 + j * 16 + n;
      const bool kvalid = kc < vlen;
#pragma unroll
      for (int r = 0; r < 8; ++r) {
        const float sb = bias_q[(size_t)r * K_ + kc];        // unconditional load
        float s = fmaf(s8[r], SCALE_L2E, sb * LOG2E);
        s = kvalid ? s : MASKVAL;                            // v_cndmask, no branch
        const float nm = fmaxf(m8[r], s);
        l8[r] = l8[r] * exp2f(m8[r] - nm) + exp2f(s - nm);
        m8[r] = nm;
      }
    }
  }
  __syncthreads();

  // combine stats across the 16-lane n-group (masked lanes vanish via exp2(m-M)=0)
#pragma unroll
  for (int r = 0; r < 8; ++r) {
    float m = m8[r], l = l8[r];
#pragma unroll
    for (int d = 1; d < 16; d <<= 1) {
      const float om = __shfl_xor(m, d, 32);
      const float ol = __shfl_xor(l, d, 32);
      const float nm = fmaxf(m, om);
      l = l * exp2f(m - nm) + ol * exp2f(om - nm);
      m = nm;
    }
    m8[r] = m;
    l8[r] = 1.0f / l;                                        // invL
  }

  // ================= Pass 2: attn out + ctx = P*V =================================
  v8f o0 = {}, o1 = {}, o2 = {}, o3 = {};
  _Float16* pw        = pbuf[wid];
  float*    attn_base = attn + (size_t)(b * H_ + h) * Q_ * K_;

  stage_k_async(khf, 0, tid, kbuf[0]);
  stage_v_async(vhf, 0, tid, vtbuf[0]);
  for (int c = 0; c < NKT; ++c) {
    wait_async0();
    __syncthreads();
    if (c + 1 < NKT) {
      stage_k_async(khf, c + 1, tid, kbuf[(c + 1) & 1]);
      stage_v_async(vhf, c + 1, tid, vtbuf[(c + 1) & 1]);
    }
    const _Float16* kb = kbuf[c & 1];
    const _Float16* vt = vtbuf[c & 1];

#pragma unroll
    for (int j = 0; j < 2; ++j) {
      v8f s8 = {};
      s8 = wmma_f16(a0, kfrag(kb, lane, j, 0),  s8);
      s8 = wmma_f16(a1, kfrag(kb, lane, j, 32), s8);
      const int  kc     = c * 32 + j * 16 + n;
      const bool kvalid = kc < vlen;
#pragma unroll
      for (int r = 0; r < 8; ++r) {
        const float sb = bias_q[(size_t)r * K_ + kc];
        float s = fmaf(s8[r], SCALE_L2E, sb * LOG2E);
        s = kvalid ? s : MASKVAL;
        const float p = exp2f(s - m8[r]) * l8[r];            // normalized probability
        attn_base[(size_t)(q0 + r + 8 * halfsel) * K_ + kc] = p;
        pw[(r + 8 * halfsel) * 32 + j * 16 + n] = (_Float16)p;
      }
    }

    // P (C-layout) -> A-fragment; DS ops are wave-ordered, region is wave-private
    v16h pa;
    {
      const _Float16* p = pw + n * 32 + halfsel * 8;
      pa = cat8(*(const v8h*)p, *(const v8h*)(p + 16));
    }

    o0 = wmma_f16(pa, vfrag(vt, lane, 0), o0);
    o1 = wmma_f16(pa, vfrag(vt, lane, 1), o1);
    o2 = wmma_f16(pa, vfrag(vt, lane, 2), o2);
    o3 = wmma_f16(pa, vfrag(vt, lane, 3), o3);
  }

  // ---- write ctx [B,Q,H,DV] ----
#pragma unroll
  for (int r = 0; r < 8; ++r) {
    const size_t base = (((size_t)(b * Q_ + q0 + r + 8 * halfsel) * H_) + h) * DV_;
    ctx[base +  0 + n] = o0[r];
    ctx[base + 16 + n] = o1[r];
    ctx[base + 32 + n] = o2[r];
    ctx[base + 48 + n] = o3[r];
  }
}

extern "C" void kernel_launch(void* const* d_in, const int* in_sizes, int n_in,
                              void* d_out, int out_size, void* d_ws, size_t ws_size,
                              hipStream_t stream) {
  (void)in_sizes; (void)n_in; (void)out_size; (void)ws_size;
  const float* qs   = (const float*)d_in[0];
  const float* ks   = (const float*)d_in[1];
  const float* vs   = (const float*)d_in[2];
  const float* bias = (const float*)d_in[3];
  const int*   vl   = (const int*)d_in[4];

  float* out  = (float*)d_out;
  float* ctx  = out;                                   // [B,Q,H,DV]
  float* attn = out + (size_t)B_ * Q_ * H_ * DV_;      // [B,H,Q,K]

  _Float16* kf16  = (_Float16*)d_ws;                   // 8 MB f16 K
  _Float16* vtf16 = kf16 + (size_t)B_ * K_ * H_ * D_;  // 8 MB f16 V^T

  const int nelem_k = B_ * K_ * H_ * D_;               // 4,194,304
  cvt_k_kernel<<<nelem_k / (256 * 8), 256, 0, stream>>>(ks, kf16);
  transpose_v_kernel<<<B_ * H_ * (K_ / 32), 256, 0, stream>>>(vs, vtf16);

  const int total_waves = B_ * H_ * (Q_ / 16);         // 4096
  attn_fwd_kernel<<<dim3(total_waves / 8), 256, 0, stream>>>(
      qs, kf16, vtf16, bias, vl, ctx, attn);
}
